// EdgeMPNNLayer_70428873720249
// MI455X (gfx1250) — matrix operand order repaired and checked by reading, not compiled, not run on hardware
//
#include <hip/hip_runtime.h>
#include <hip/hip_bf16.h>

typedef __attribute__((ext_vector_type(16))) __bf16 v16bf;
typedef __attribute__((ext_vector_type(8)))  __bf16 v8bf;
typedef __attribute__((ext_vector_type(4)))  __bf16 v4bf;
typedef __attribute__((ext_vector_type(8)))  float  v8f;
typedef __attribute__((ext_vector_type(4)))  float  v4f;

#define HDIM 256
#define EDIM 64
#define KIN_E 576          // 2*HDIM + EDIM
#define MSGH 512
#define UPDH 512
#define MTILE 32           // edges/nodes per block (2 WMMA M-tiles)

// ---------------------------------------------------------------------------
__global__ void zero_f32(float* p, int n) {
    int i = blockIdx.x * blockDim.x + threadIdx.x;
    if (i < n) p[i] = 0.0f;
}

// ---------------------------------------------------------------------------
// Repack f32 weight W[K][N] (row-major, x @ W) into bf16 WMMA-B tiles.
// tile t = nt*tilesK + kt holds 32 lanes * 16 bf16:
//   lane n, elem j  ->  W[kt*32 + (j>=8?16:0) + (n>=16?8:0) + (j&7)][nt*16 + (n&15)]
// = CDNA5 16-bit B-operand (32x16) per-lane register order.
// ---------------------------------------------------------------------------
__global__ void pack_w(const float* __restrict__ W, __bf16* __restrict__ Wp,
                       int K, int N) {
    int idx = blockIdx.x * blockDim.x + threadIdx.x;
    if (idx >= K * N) return;
    int tilesK = K >> 5;
    int j    = idx & 15;
    int lane = (idx >> 4) & 31;
    int t    = idx >> 9;
    int kt   = t % tilesK;
    int nt   = t / tilesK;
    int k = kt * 32 + ((j >> 3) << 4) + ((lane >> 4) << 3) + (j & 7);
    int n = nt * 16 + (lane & 15);
    Wp[idx] = (__bf16)W[(size_t)k * N + n];
}

// ---------------------------------------------------------------------------
// 16x32 bf16 A-tile from LDS (row-major) in CDNA5 A-operand order:
//   lane<16 : M=lane,    K = kb+0..7  and kb+16..23
//   lane>=16: M=lane-16, K = kb+8..15 and kb+24..31
// ---------------------------------------------------------------------------
__device__ inline v16bf load_a_tile(const __bf16* lds, int stride, int lane, int kb) {
    int row  = lane & 15;
    int half = (lane >> 4) << 3;
    const __bf16* p = lds + row * stride + kb + half;
    v8bf lo = *(const v8bf*)(p);
    v8bf hi = *(const v8bf*)(p + 16);
    return __builtin_shufflevector(lo, hi, 0,1,2,3,4,5,6,7,8,9,10,11,12,13,14,15);
}

__device__ inline v8f bias_init(const float* b, int col) {
    float v = b[col];
    v8f c;
#pragma unroll
    for (int i = 0; i < 8; ++i) c[i] = v;
    return c;
}

__device__ inline v4bf cvt4(v4f f) {
    v4bf o;
#pragma unroll
    for (int i = 0; i < 4; ++i) o[i] = (__bf16)f[i];
    return o;
}

// ---------------------------------------------------------------------------
// Edge kernel: block = 32 edges, 256 threads (8 waves).
//  x   = [h[src], h[dst], ea]   (32 x 576 bf16 in LDS)
//  hid = relu(x @ W1 + b1)      (32 x 512 bf16 in LDS)
//  m   = hid @ W2 + b2          (32 x 256 f32 in regs)
//  agg[dst] += m                (hw f32 atomics)
// Each wave: layer1 = 2 M-tiles x 4 N-tiles (B reused across M),
//            layer2 = 2 M-tiles x 2 N-tiles.
// ---------------------------------------------------------------------------
__global__ __launch_bounds__(256)
void edge_kernel(const float* __restrict__ h,
                 const long long* __restrict__ ei,   // [2, E] int64
                 const float* __restrict__ ea,
                 const __bf16* __restrict__ W1p, const float* __restrict__ b1,
                 const __bf16* __restrict__ W2p, const float* __restrict__ b2,
                 float* __restrict__ agg, int nEdges) {
    __shared__ __bf16 xin[MTILE * KIN_E];
    __shared__ __bf16 hid[MTILE * MSGH];
    __shared__ int    srcs[MTILE];
    __shared__ int    dsts[MTILE];

    const int tid = threadIdx.x;
    const int e0  = blockIdx.x * MTILE;
    const long long* srcI = ei;
    const long long* dstI = ei + nEdges;

    if (tid < MTILE) {
        long long e = e0 + tid;
        srcs[tid] = (e < nEdges) ? (int)srcI[e] : 0;
        dsts[tid] = (e < nEdges) ? (int)dstI[e] : 0;
    }
    __syncthreads();

    // vectorized gather: float4 loads -> 4x bf16 LDS stores
    for (int q = tid; q < MTILE * (KIN_E / 4); q += 256) {
        int r  = q / (KIN_E / 4);
        int c  = (q - r * (KIN_E / 4)) * 4;
        v4f v = (v4f){0.f, 0.f, 0.f, 0.f};
        if (e0 + r < nEdges) {
            if (c < HDIM)          v = *(const v4f*)(h  + (size_t)srcs[r] * HDIM + c);
            else if (c < 2 * HDIM) v = *(const v4f*)(h  + (size_t)dsts[r] * HDIM + (c - HDIM));
            else                   v = *(const v4f*)(ea + (size_t)(e0 + r) * EDIM + (c - 2 * HDIM));
        }
        *(v4bf*)(&xin[r * KIN_E + c]) = cvt4(v);
    }
    __syncthreads();

    const int lane  = tid & 31;
    const int wave  = tid >> 5;
    const int col16 = lane & 15;
    const int rbase = (lane >> 4) << 3;

    // ---- layer 1: [32,576] x [576,512]; wave: 2M x 4N, 18 K-tiles
    {
        v8f acc[2][4];
        const __bf16* wb[4];
#pragma unroll
        for (int t = 0; t < 4; ++t) {
            int nt = wave * 4 + t;
            acc[0][t] = bias_init(b1, nt * 16 + col16);
            acc[1][t] = acc[0][t];
            wb[t] = W1p + (size_t)nt * (KIN_E / 32) * 512 + lane * 16;
        }
#pragma unroll
        for (int kt = 0; kt < KIN_E / 32; ++kt) {
            v16bf a0 = load_a_tile(xin,              KIN_E, lane, kt * 32);
            v16bf a1 = load_a_tile(xin + 16 * KIN_E, KIN_E, lane, kt * 32);
#pragma unroll
            for (int t = 0; t < 4; ++t) {
                v16bf bm = *(const v16bf*)(wb[t] + kt * 512);
                acc[0][t] = __builtin_amdgcn_wmma_f32_16x16x32_bf16(
                    false, a0, false, bm, (short)0, acc[0][t], false, false);
                acc[1][t] = __builtin_amdgcn_wmma_f32_16x16x32_bf16(
                    false, a1, false, bm, (short)0, acc[1][t], false, false);
            }
        }
#pragma unroll
        for (int t = 0; t < 4; ++t) {
            int col = (wave * 4 + t) * 16 + col16;
#pragma unroll
            for (int m = 0; m < 2; ++m)
#pragma unroll
                for (int i = 0; i < 8; ++i)
                    hid[(m * 16 + rbase + i) * MSGH + col] =
                        (__bf16)fmaxf(acc[m][t][i], 0.0f);
        }
    }
    __syncthreads();

    // ---- layer 2: [32,512] x [512,256]; wave: 2M x 2N, 16 K-tiles
    {
        v8f acc[2][2];
        const __bf16* wb[2];
#pragma unroll
        for (int t = 0; t < 2; ++t) {
            int nt = wave * 2 + t;
            acc[0][t] = bias_init(b2, nt * 16 + col16);
            acc[1][t] = acc[0][t];
            wb[t] = W2p + (size_t)nt * (MSGH / 32) * 512 + lane * 16;
        }
#pragma unroll
        for (int kt = 0; kt < MSGH / 32; ++kt) {
            v16bf a0 = load_a_tile(hid,             MSGH, lane, kt * 32);
            v16bf a1 = load_a_tile(hid + 16 * MSGH, MSGH, lane, kt * 32);
#pragma unroll
            for (int t = 0; t < 2; ++t) {
                v16bf bm = *(const v16bf*)(wb[t] + kt * 512);
                acc[0][t] = __builtin_amdgcn_wmma_f32_16x16x32_bf16(
                    false, a0, false, bm, (short)0, acc[0][t], false, false);
                acc[1][t] = __builtin_amdgcn_wmma_f32_16x16x32_bf16(
                    false, a1, false, bm, (short)0, acc[1][t], false, false);
            }
        }
#pragma unroll
        for (int t = 0; t < 2; ++t) {
            int col = (wave * 2 + t) * 16 + col16;
#pragma unroll
            for (int m = 0; m < 2; ++m)
#pragma unroll
                for (int i = 0; i < 8; ++i) {
                    int row = m * 16 + rbase + i;
                    if (e0 + row < nEdges)
                        unsafeAtomicAdd(&agg[(size_t)dsts[row] * HDIM + col],
                                        acc[m][t][i]);
                }
        }
    }
}

// ---------------------------------------------------------------------------
// Node kernel: block = 32 nodes.
//  x = [h, agg] (32x512), hid = relu(x@nW1+b1), u = hid@nW2+b2
//  out = LayerNorm(h + u) * gamma + beta
// ---------------------------------------------------------------------------
__global__ __launch_bounds__(256)
void node_kernel(const float* __restrict__ h, const float* __restrict__ agg,
                 const __bf16* __restrict__ W1p, const float* __restrict__ b1,
                 const __bf16* __restrict__ W2p, const float* __restrict__ b2,
                 const float* __restrict__ gamma, const float* __restrict__ beta,
                 float* __restrict__ out, int nNodes) {
    __shared__ __bf16 xin[MTILE * 2 * HDIM];
    __shared__ __bf16 hid[MTILE * UPDH];
    __shared__ float  ob[MTILE * HDIM];

    const int tid = threadIdx.x;
    const int n0  = blockIdx.x * MTILE;

    for (int q = tid; q < MTILE * (2 * HDIM / 4); q += 256) {
        int r = q >> 7;              // 512/4 = 128 quads per row
        int c = (q & 127) * 4;
        v4f v = (v4f){0.f, 0.f, 0.f, 0.f};
        if (n0 + r < nNodes)
            v = (c < HDIM) ? *(const v4f*)(h   + (size_t)(n0 + r) * HDIM + c)
                           : *(const v4f*)(agg + (size_t)(n0 + r) * HDIM + (c - HDIM));
        *(v4bf*)(&xin[r * (2 * HDIM) + c]) = cvt4(v);
    }
    __syncthreads();

    const int lane  = tid & 31;
    const int wave  = tid >> 5;
    const int col16 = lane & 15;
    const int rbase = (lane >> 4) << 3;

    // ---- layer 1: [32,512] x [512,512]; wave: 2M x 4N, 16 K-tiles
    {
        v8f acc[2][4];
        const __bf16* wb[4];
#pragma unroll
        for (int t = 0; t < 4; ++t) {
            int nt = wave * 4 + t;
            acc[0][t] = bias_init(b1, nt * 16 + col16);
            acc[1][t] = acc[0][t];
            wb[t] = W1p + (size_t)nt * ((2 * HDIM) / 32) * 512 + lane * 16;
        }
#pragma unroll
        for (int kt = 0; kt < (2 * HDIM) / 32; ++kt) {
            v16bf a0 = load_a_tile(xin,                  2 * HDIM, lane, kt * 32);
            v16bf a1 = load_a_tile(xin + 16 * 2 * HDIM,  2 * HDIM, lane, kt * 32);
#pragma unroll
            for (int t = 0; t < 4; ++t) {
                v16bf bm = *(const v16bf*)(wb[t] + kt * 512);
                acc[0][t] = __builtin_amdgcn_wmma_f32_16x16x32_bf16(
                    false, a0, false, bm, (short)0, acc[0][t], false, false);
                acc[1][t] = __builtin_amdgcn_wmma_f32_16x16x32_bf16(
                    false, a1, false, bm, (short)0, acc[1][t], false, false);
            }
        }
#pragma unroll
        for (int t = 0; t < 4; ++t) {
            int col = (wave * 4 + t) * 16 + col16;
#pragma unroll
            for (int m = 0; m < 2; ++m)
#pragma unroll
                for (int i = 0; i < 8; ++i)
                    hid[(m * 16 + rbase + i) * UPDH + col] =
                        (__bf16)fmaxf(acc[m][t][i], 0.0f);
        }
    }
    __syncthreads();

    // ---- layer 2: [32,512] x [512,256]; wave: 2M x 2N, 16 K-tiles
    {
        v8f acc[2][2];
        const __bf16* wb[2];
#pragma unroll
        for (int t = 0; t < 2; ++t) {
            int nt = wave * 2 + t;
            acc[0][t] = bias_init(b2, nt * 16 + col16);
            acc[1][t] = acc[0][t];
            wb[t] = W2p + (size_t)nt * (UPDH / 32) * 512 + lane * 16;
        }
#pragma unroll
        for (int kt = 0; kt < UPDH / 32; ++kt) {
            v16bf a0 = load_a_tile(hid,             UPDH, lane, kt * 32);
            v16bf a1 = load_a_tile(hid + 16 * UPDH, UPDH, lane, kt * 32);
#pragma unroll
            for (int t = 0; t < 2; ++t) {
                v16bf bm = *(const v16bf*)(wb[t] + kt * 512);
                acc[0][t] = __builtin_amdgcn_wmma_f32_16x16x32_bf16(
                    false, a0, false, bm, (short)0, acc[0][t], false, false);
                acc[1][t] = __builtin_amdgcn_wmma_f32_16x16x32_bf16(
                    false, a1, false, bm, (short)0, acc[1][t], false, false);
            }
        }
#pragma unroll
        for (int t = 0; t < 2; ++t) {
            int col = (wave * 2 + t) * 16 + col16;
#pragma unroll
            for (int m = 0; m < 2; ++m)
#pragma unroll
                for (int i = 0; i < 8; ++i)
                    ob[(m * 16 + rbase + i) * HDIM + col] = acc[m][t][i];
        }
    }
    __syncthreads();

    // ---- residual + LayerNorm: 8 lanes per row, 32 cols each (stride 8)
    {
        int r  = tid >> 3;          // 0..31
        int t8 = tid & 7;
        float vals[32];
        float s = 0.0f, ss = 0.0f;
        size_t rowBase = (size_t)(n0 + r) * HDIM;
        bool valid = (n0 + r) < nNodes;
#pragma unroll
        for (int i = 0; i < 32; ++i) {
            int c = t8 + i * 8;
            float v = valid ? (h[rowBase + c] + ob[r * HDIM + c]) : 0.0f;
            vals[i] = v;
            s += v; ss += v * v;
        }
        for (int m = 1; m < 8; m <<= 1) {           // butterfly within 8-lane group
            s  += __shfl_xor(s,  m, 32);
            ss += __shfl_xor(ss, m, 32);
        }
        float mu  = s * (1.0f / HDIM);
        float var = ss * (1.0f / HDIM) - mu * mu;
        float inv = rsqrtf(var + 1e-5f);
        if (valid) {
#pragma unroll
            for (int i = 0; i < 32; ++i) {
                int c = t8 + i * 8;
                out[rowBase + c] = (vals[i] - mu) * inv * gamma[c] + beta[c];
            }
        }
    }
}

// ---------------------------------------------------------------------------
extern "C" void kernel_launch(void* const* d_in, const int* in_sizes, int n_in,
                              void* d_out, int out_size, void* d_ws, size_t ws_size,
                              hipStream_t stream) {
    const float*     h   = (const float*)d_in[0];
    const long long* ei  = (const long long*)d_in[1];   // int64 [2,E]
    const float*     ea  = (const float*)d_in[2];
    const float*     eW1 = (const float*)d_in[3];
    const float*     eb1 = (const float*)d_in[4];
    const float*     eW2 = (const float*)d_in[5];
    const float*     eb2 = (const float*)d_in[6];
    const float*     nW1 = (const float*)d_in[7];
    const float*     nb1 = (const float*)d_in[8];
    const float*     nW2 = (const float*)d_in[9];
    const float*     nb2 = (const float*)d_in[10];
    const float*     gam = (const float*)d_in[11];
    const float*     bet = (const float*)d_in[12];
    float* out = (float*)d_out;

    const int nNodes = in_sizes[0] / HDIM;
    const int nEdges = in_sizes[1] / 2;

    // workspace: agg | eW1p | eW2p | nW1p | nW2p
    char* ws = (char*)d_ws;
    size_t off = 0;
    float* agg = (float*)(ws + off);  off += (size_t)nNodes * HDIM * sizeof(float);
    off = (off + 255) & ~(size_t)255;
    __bf16* eW1p = (__bf16*)(ws + off); off += (size_t)KIN_E * MSGH * 2;
    off = (off + 255) & ~(size_t)255;
    __bf16* eW2p = (__bf16*)(ws + off); off += (size_t)MSGH * HDIM * 2;
    off = (off + 255) & ~(size_t)255;
    __bf16* nW1p = (__bf16*)(ws + off); off += (size_t)(2 * HDIM) * UPDH * 2;
    off = (off + 255) & ~(size_t)255;
    __bf16* nW2p = (__bf16*)(ws + off); off += (size_t)UPDH * HDIM * 2;

    int aggN = nNodes * HDIM;
    zero_f32<<<(aggN + 255) / 256, 256, 0, stream>>>(agg, aggN);

    pack_w<<<(KIN_E * MSGH + 255) / 256, 256, 0, stream>>>(eW1, eW1p, KIN_E, MSGH);
    pack_w<<<(MSGH * HDIM + 255) / 256, 256, 0, stream>>>(eW2, eW2p, MSGH, HDIM);
    pack_w<<<(2 * HDIM * UPDH + 255) / 256, 256, 0, stream>>>(nW1, nW1p, 2 * HDIM, UPDH);
    pack_w<<<(UPDH * HDIM + 255) / 256, 256, 0, stream>>>(nW2, nW2p, UPDH, HDIM);

    int eBlocks = (nEdges + MTILE - 1) / MTILE;
    edge_kernel<<<eBlocks, 256, 0, stream>>>(h, ei, ea, eW1p, eb1, eW2p, eb2, agg, nEdges);

    int nBlocks = (nNodes + MTILE - 1) / MTILE;
    node_kernel<<<nBlocks, 256, 0, stream>>>(h, agg, nW1p, nb1, nW2p, nb2, gam, bet, out, nNodes);
}
